// StandardSoftmaxAttention_81853486727564
// MI455X (gfx1250) — matrix operand order repaired and checked by reading, not compiled, not run on hardware
//
#include <hip/hip_runtime.h>
#include <hip/hip_bf16.h>
#include <cstdint>

// ---------------------------------------------------------------------------
// Multi-head causal attention block for MI455X (gfx1250, wave32, WMMA).
//   y = OutProj( Softmax( (xWq)(xWk)^T / sqrt(hd) , causal ) (xWv) )
// Matrix math on v_wmma_f32_16x16x32_f16; GEMM tile movement on the CDNA5
// async DMA path (global_load_async_to_lds_b128, ASYNCcnt-gated, double
// buffered in LDS).
// ---------------------------------------------------------------------------

typedef __attribute__((ext_vector_type(16))) _Float16 v16h;
typedef __attribute__((ext_vector_type(8)))  float    v8f;
typedef __attribute__((ext_vector_type(4)))  _Float16 v4h;

union F16Frag { v16h v; uint4 q[2]; };

__device__ __forceinline__ v8f wmma_f16(v16h a, v16h b, v8f c) {
  // D = A(16x32 f16) * B(32x16 f16) + C(16x16 f32)
  return __builtin_amdgcn_wmma_f32_16x16x32_f16(false, a, false, b, (short)0, c,
                                                false, false);
}

// Async DMA: 16 bytes global -> LDS, tracked by ASYNCcnt (one per wave-inst).
__device__ __forceinline__ void async_copy16(_Float16* dst_lds,
                                             const _Float16* src) {
  // generic -> addrspace(3) cast so the VGPR holds the real LDS byte offset
  uint32_t l = (uint32_t)(uintptr_t)(__attribute__((address_space(3))) _Float16*)dst_lds;
  asm volatile("global_load_async_to_lds_b128 %0, %1, off"
               :: "v"(l), "v"((uint64_t)(uintptr_t)src)
               : "memory");
}

#define EMBED 1024
#define NH    16
#define HD    64
#define BATCH 2
#define SEQ   2048
#define BT    (BATCH * SEQ)   // 4096 tokens
#define NQKV  (3 * EMBED)     // 3072
#define LDSW  72              // 64 + 8 pad halves: 144B rows, 16B aligned

// ---------------------------------------------------------------------------
// fp32 -> f16 conversion (vectorized 4-wide)
// ---------------------------------------------------------------------------
__global__ void cvt_f32_to_f16(const float* __restrict__ src,
                               _Float16* __restrict__ dst, int n4) {
  int i = blockIdx.x * blockDim.x + threadIdx.x;
  if (i < n4) {
    float4 f = ((const float4*)src)[i];
    v4h h;
    h.x = (_Float16)f.x; h.y = (_Float16)f.y;
    h.z = (_Float16)f.z; h.w = (_Float16)f.w;
    ((v4h*)dst)[i] = h;
  }
}

// ---------------------------------------------------------------------------
// Shared 128x128 GEMM mainloop: C = A * B^T, A is Mx1024, B is Nx1024 (both
// row-major f16, K=1024).  Double-buffered async-DMA LDS staging, K-step 64,
// 8 waves of 64x32, 16 WMMAs per stage.  Result left in acc[4][2] (C-layout).
// ---------------------------------------------------------------------------
__device__ __forceinline__ void gemm_mainloop_128x128(
    const _Float16* __restrict__ Ag,   // + blockM row offset applied by caller
    const _Float16* __restrict__ Bg,   // + blockN row offset applied by caller
    _Float16 (&As)[2][128][LDSW], _Float16 (&Bs)[2][128][LDSW],
    v8f (&acc)[4][2]) {
  const int t    = threadIdx.x;
  const int lane = t & 31, wid = t >> 5;
  const int lm   = lane & 15, lh = lane >> 4;
  const int waveM = (wid >> 2) * 64;
  const int waveN = (wid & 3) * 32;

  const int lr = t >> 1;            // 0..127: slab row this thread DMAs
  const int lc = (t & 1) * 32;      // 0/32 halves: 64B half-row

  auto issue_tile = [&](int buf, int k0) {
    const _Float16* ga = Ag + (size_t)lr * EMBED + k0 + lc;
    const _Float16* gb = Bg + (size_t)lr * EMBED + k0 + lc;
#pragma unroll
    for (int j = 0; j < 4; ++j) {            // 4 x 16B per matrix per thread
      async_copy16(&As[buf][lr][lc + 8 * j], ga + 8 * j);
      async_copy16(&Bs[buf][lr][lc + 8 * j], gb + 8 * j);
    }
  };

  issue_tile(0, 0);

  for (int k0 = 0; k0 < EMBED; k0 += 64) {
    const int cur = (k0 >> 6) & 1;
    if (k0 + 64 < EMBED) {
      issue_tile(cur ^ 1, k0 + 64);          // prefetch next slab
      // 8 newly-issued ops outstanding; async loads complete in order, so
      // cnt<=8 proves the previous slab's 8 ops have landed in LDS.
      asm volatile("s_wait_asynccnt 0x8" ::: "memory");
    } else {
      asm volatile("s_wait_asynccnt 0x0" ::: "memory");
    }
    __syncthreads();                         // slab visible to all waves

#pragma unroll
    for (int kc = 0; kc < 2; ++kc) {         // two K=32 chunks per slab
      F16Frag a[4], b[2];
#pragma unroll
      for (int mt = 0; mt < 4; ++mt) {       // A frag: row=lm, K per lane-half
        const _Float16* p = &As[cur][waveM + mt * 16 + lm][32 * kc + lh * 16];
        a[mt].q[0] = *(const uint4*)p;
        a[mt].q[1] = *(const uint4*)(p + 8);
      }
#pragma unroll
      for (int nt = 0; nt < 2; ++nt) {       // B frag: col=lm, K = lh*16 + i
        const _Float16* p = &Bs[cur][waveN + nt * 16 + lm][32 * kc + lh * 16];
        b[nt].q[0] = *(const uint4*)p;
        b[nt].q[1] = *(const uint4*)(p + 8);
      }
#pragma unroll
      for (int mt = 0; mt < 4; ++mt)
#pragma unroll
        for (int nt = 0; nt < 2; ++nt)
          acc[mt][nt] = wmma_f16(a[mt].v, b[nt].v, acc[mt][nt]);
    }
    __syncthreads();   // all waves done reading before this slab is re-filled
  }
}

// ---------------------------------------------------------------------------
// QKV projection GEMM: C[4096 x 3072] = Xh * Wqkv^T. Epilogue scatters into
// Q (pre-scaled by 1/sqrt(hd)), K as (B,H,T,hd), V transposed as (B,H,hd,T).
// ---------------------------------------------------------------------------
__launch_bounds__(256)
__global__ void qkv_gemm(const _Float16* __restrict__ Xh,
                         const _Float16* __restrict__ Wh,
                         _Float16* __restrict__ Qw,
                         _Float16* __restrict__ Kw,
                         _Float16* __restrict__ VTw) {
  __shared__ _Float16 As[2][128][LDSW];
  __shared__ _Float16 Bs[2][128][LDSW];

  const int t    = threadIdx.x;
  const int lane = t & 31, wid = t >> 5;
  const int lm   = lane & 15, lh = lane >> 4;
  const int blockM = blockIdx.y * 128;
  const int blockN = blockIdx.x * 128;
  const int waveM  = (wid >> 2) * 64;
  const int waveN  = (wid & 3) * 32;

  v8f zero = {};
  v8f acc[4][2];
#pragma unroll
  for (int mt = 0; mt < 4; ++mt)
#pragma unroll
    for (int nt = 0; nt < 2; ++nt) acc[mt][nt] = zero;

  gemm_mainloop_128x128(Xh + (size_t)blockM * EMBED,
                        Wh + (size_t)blockN * EMBED, As, Bs, acc);

  // Scatter epilogue: C-layout element i of lane l is (row i+8*lh, col lm).
#pragma unroll
  for (int mt = 0; mt < 4; ++mt)
#pragma unroll
    for (int nt = 0; nt < 2; ++nt)
#pragma unroll
      for (int i = 0; i < 8; ++i) {
        int row = blockM + waveM + mt * 16 + i + 8 * lh;  // token 0..4095
        int col = blockN + waveN + nt * 16 + lm;          // e     0..3071
        float val = acc[mt][nt][i];
        int bb = row >> 11, tt = row & 2047;
        int three = col >> 10, h = (col >> 6) & 15, d = col & 63;
        size_t bh = (size_t)(bb * NH + h);
        if (three == 0)
          Qw[(bh * SEQ + tt) * HD + d] = (_Float16)(val * 0.125f); // 1/sqrt(64)
        else if (three == 1)
          Kw[(bh * SEQ + tt) * HD + d] = (_Float16)val;
        else
          VTw[(bh * HD + d) * SEQ + tt] = (_Float16)val;           // transposed
      }
}

// ---------------------------------------------------------------------------
// Output projection GEMM: out[4096 x 1024] = Ah * Wout^T, fp32 result.
// ---------------------------------------------------------------------------
__launch_bounds__(256)
__global__ void out_gemm(const _Float16* __restrict__ Ah,
                         const _Float16* __restrict__ Wh,
                         float* __restrict__ Out) {
  __shared__ _Float16 As[2][128][LDSW];
  __shared__ _Float16 Bs[2][128][LDSW];

  const int t    = threadIdx.x;
  const int lane = t & 31, wid = t >> 5;
  const int lm   = lane & 15, lh = lane >> 4;
  const int blockM = blockIdx.y * 128;
  const int blockN = blockIdx.x * 128;
  const int waveM  = (wid >> 2) * 64;
  const int waveN  = (wid & 3) * 32;

  v8f zero = {};
  v8f acc[4][2];
#pragma unroll
  for (int mt = 0; mt < 4; ++mt)
#pragma unroll
    for (int nt = 0; nt < 2; ++nt) acc[mt][nt] = zero;

  gemm_mainloop_128x128(Ah + (size_t)blockM * EMBED,
                        Wh + (size_t)blockN * EMBED, As, Bs, acc);

#pragma unroll
  for (int mt = 0; mt < 4; ++mt)
#pragma unroll
    for (int nt = 0; nt < 2; ++nt)
#pragma unroll
      for (int i = 0; i < 8; ++i) {
        int row = blockM + waveM + mt * 16 + i + 8 * lh;
        int col = blockN + waveN + nt * 16 + lm;
        Out[(size_t)row * EMBED + col] = acc[mt][nt][i];
      }
}

// ---------------------------------------------------------------------------
// Flash attention: one wave per 16-row query tile of one (b,h).
// Online softmax; score/PV tiles via WMMA; 32 keys per step.
// ---------------------------------------------------------------------------
__launch_bounds__(128)
__global__ void flash_attn(const _Float16* __restrict__ Q,
                           const _Float16* __restrict__ K,
                           const _Float16* __restrict__ VT,
                           _Float16* __restrict__ Aout) {
  __shared__ _Float16 Pl[4][16][40];   // per-wave P staging (C-layout -> A-frag)

  const int t    = threadIdx.x;
  const int lane = t & 31, wid = t >> 5;
  const int lm   = lane & 15, lh = lane >> 4;
  const int gw = blockIdx.x * 4 + wid;      // 0..4095 waves
  const int qt = gw & 127;                  // query tile within sequence
  const int bh = gw >> 7;                   // (b*16 + h) 0..31
  const int q0 = qt * 16;

  const _Float16* Qb = Q + (size_t)bh * SEQ * HD;
  const _Float16* Kb = K + (size_t)bh * SEQ * HD;
  const _Float16* Vb = VT + (size_t)bh * HD * SEQ;

  // Q A-fragments for hd-chunks 0..31 and 32..63 (loaded once).
  F16Frag qa[2];
#pragma unroll
  for (int c = 0; c < 2; ++c) {
    const _Float16* p = Qb + (size_t)(q0 + lm) * HD + 32 * c + lh * 16;
    qa[c].q[0] = *(const uint4*)p;
    qa[c].q[1] = *(const uint4*)(p + 8);
  }

  v8f zero = {};
  v8f o[4];                          // O accum, 16 x 64 fp32 (4 N-tiles of hd)
  float mrun[8], lrun[8];
#pragma unroll
  for (int ht = 0; ht < 4; ++ht) o[ht] = zero;
#pragma unroll
  for (int i = 0; i < 8; ++i) { mrun[i] = -__builtin_inff(); lrun[i] = 0.f; }

  for (int key0 = 0; key0 <= q0 + 15; key0 += 32) {
    // ---- S = Q K^T for 32 keys (two 16-key N tiles) ----
    v8f s0 = zero, s1 = zero;
#pragma unroll
    for (int nt = 0; nt < 2; ++nt) {
      F16Frag kb0, kb1;               // B-frag from K rows: contiguous in hd
      const _Float16* p = Kb + (size_t)(key0 + nt * 16 + lm) * HD + lh * 16;
      kb0.q[0] = *(const uint4*)p;      kb0.q[1] = *(const uint4*)(p + 8);
      kb1.q[0] = *(const uint4*)(p+32); kb1.q[1] = *(const uint4*)(p + 40);
      if (nt == 0) { s0 = wmma_f16(qa[0].v, kb0.v, s0); s0 = wmma_f16(qa[1].v, kb1.v, s0); }
      else         { s1 = wmma_f16(qa[0].v, kb0.v, s1); s1 = wmma_f16(qa[1].v, kb1.v, s1); }
    }

    // ---- causal mask + row max (reduce across the 16 N-lanes) ----
    const float NEG = -__builtin_inff();
    float rmax[8];
#pragma unroll
    for (int i = 0; i < 8; ++i) {
      int qrow = q0 + i + 8 * lh;
      if (key0 + lm > qrow)      s0[i] = NEG;
      if (key0 + 16 + lm > qrow) s1[i] = NEG;
      rmax[i] = fmaxf(s0[i], s1[i]);
    }
#pragma unroll
    for (int m = 1; m < 16; m <<= 1)
#pragma unroll
      for (int i = 0; i < 8; ++i)
        rmax[i] = fmaxf(rmax[i], __shfl_xor(rmax[i], m, 32));

    float mnew[8], scale[8], rsum[8];
#pragma unroll
    for (int i = 0; i < 8; ++i) {
      mnew[i]  = fmaxf(mrun[i], rmax[i]);     // finite: diagonal always live
      scale[i] = __expf(mrun[i] - mnew[i]);   // exp(-inf)=0 on first step
    }
#pragma unroll
    for (int ht = 0; ht < 4; ++ht)
#pragma unroll
      for (int i = 0; i < 8; ++i) o[ht][i] *= scale[i];

    // ---- P = exp(S - mnew): stage to LDS (C-layout -> A-frag re-layout) ----
#pragma unroll
    for (int i = 0; i < 8; ++i) {
      float p0 = __expf(s0[i] - mnew[i]);
      float p1 = __expf(s1[i] - mnew[i]);
      Pl[wid][i + 8 * lh][lm]      = (_Float16)p0;
      Pl[wid][i + 8 * lh][lm + 16] = (_Float16)p1;
      rsum[i] = p0 + p1;
    }
#pragma unroll
    for (int m = 1; m < 16; m <<= 1)
#pragma unroll
      for (int i = 0; i < 8; ++i) rsum[i] += __shfl_xor(rsum[i], m, 32);
#pragma unroll
    for (int i = 0; i < 8; ++i) {
      lrun[i] = lrun[i] * scale[i] + rsum[i];
      mrun[i] = mnew[i];
    }

    // wave-private LDS: drain DS stores before cross-lane reload
    asm volatile("s_wait_dscnt 0x0" ::: "memory");

    F16Frag pa;                     // P as A-frag (16 rows x 32 keys)
    {
      const _Float16* p = &Pl[wid][lm][lh * 16];
      pa.q[0] = *(const uint4*)p;
      pa.q[1] = *(const uint4*)(p + 8);
    }
    // ---- O += P V : V^T layout makes B-frags contiguous per lane ----
#pragma unroll
    for (int ht = 0; ht < 4; ++ht) {
      F16Frag vb;
      const _Float16* p = Vb + (size_t)(ht * 16 + lm) * SEQ + key0 + lh * 16;
      vb.q[0] = *(const uint4*)p;
      vb.q[1] = *(const uint4*)(p + 8);
      o[ht] = wmma_f16(pa.v, vb.v, o[ht]);
    }
  }

  // ---- normalize and write attention output as (b, t, h*64+d) f16 ----
  float inv[8];
#pragma unroll
  for (int i = 0; i < 8; ++i) inv[i] = 1.0f / lrun[i];
  const int bb = bh >> 4, h = bh & 15;
#pragma unroll
  for (int ht = 0; ht < 4; ++ht)
#pragma unroll
    for (int i = 0; i < 8; ++i) {
      int tt = q0 + i + 8 * lh;
      int d  = h * HD + ht * 16 + lm;
      Aout[((size_t)bb * SEQ + tt) * EMBED + d] = (_Float16)(o[ht][i] * inv[i]);
    }
}

// ---------------------------------------------------------------------------
extern "C" void kernel_launch(void* const* d_in, const int* in_sizes, int n_in,
                              void* d_out, int out_size, void* d_ws, size_t ws_size,
                              hipStream_t stream) {
  (void)in_sizes; (void)n_in; (void)out_size; (void)ws_size;
  const float* x     = (const float*)d_in[0];   // (B,T,D) fp32
  const float* qkv_w = (const float*)d_in[1];   // (3D,D)  fp32
  const float* out_w = (const float*)d_in[2];   // (D,D)   fp32
  float* out = (float*)d_out;                   // (B,T,D) fp32

  // Workspace carve-up (all f16): ~48 MB total
  char* w = (char*)d_ws;
  _Float16* Xh    = (_Float16*)w; w += (size_t)BT * EMBED * 2;
  _Float16* Wqkvh = (_Float16*)w; w += (size_t)NQKV * EMBED * 2;
  _Float16* Wouth = (_Float16*)w; w += (size_t)EMBED * EMBED * 2;
  _Float16* Qw    = (_Float16*)w; w += (size_t)BT * EMBED * 2;   // (B,H,T,hd)
  _Float16* Kw    = (_Float16*)w; w += (size_t)BT * EMBED * 2;   // (B,H,T,hd)
  _Float16* VTw   = (_Float16*)w; w += (size_t)BT * EMBED * 2;   // (B,H,hd,T)
  _Float16* Ah    = (_Float16*)w; w += (size_t)BT * EMBED * 2;   // (B,T,D)

  int n4;
  n4 = BT * EMBED / 4;
  cvt_f32_to_f16<<<(n4 + 255) / 256, 256, 0, stream>>>(x, Xh, n4);
  n4 = NQKV * EMBED / 4;
  cvt_f32_to_f16<<<(n4 + 255) / 256, 256, 0, stream>>>(qkv_w, Wqkvh, n4);
  n4 = EMBED * EMBED / 4;
  cvt_f32_to_f16<<<(n4 + 255) / 256, 256, 0, stream>>>(out_w, Wouth, n4);

  qkv_gemm<<<dim3(NQKV / 128, BT / 128), 256, 0, stream>>>(Xh, Wqkvh, Qw, Kw, VTw);
  flash_attn<<<(BATCH * NH * (SEQ / 16)) / 4, 128, 0, stream>>>(Qw, Kw, VTw, Ah);
  out_gemm<<<dim3(EMBED / 128, BT / 128), 256, 0, stream>>>(Ah, Wouth, out);
}
